// MUNEGC_36764920054021
// MI455X (gfx1250) — matrix operand order repaired and checked by reading, not compiled, not run on hardware
//
#include <hip/hip_runtime.h>
#include <hip/hip_bf16.h>

typedef __attribute__((ext_vector_type(16))) _Float16 v16h;
typedef __attribute__((ext_vector_type(8)))  _Float16 v8h;
typedef __attribute__((ext_vector_type(8)))  float    v8f;

#define NODES   4096
#define KNN     9
#define INC     32
#define OUTC    64
#define HID     128
#define EDGES   (NODES * KNN)       // 36864
#define KDIM    (HID * INC)         // 4096 (GEMM K dimension)
#define WGWAVES 8
#define EPW     (WGWAVES * 16)      // 128 edges per workgroup

__device__ __forceinline__ void wait_asynccnt0() {
#if __has_builtin(__builtin_amdgcn_s_wait_asynccnt)
    __builtin_amdgcn_s_wait_asynccnt(0);
#else
    asm volatile("s_wait_asynccnt 0x0" ::: "memory");
#endif
}

// Async global -> LDS copy of 16 bytes per lane (CDNA5, ASYNCcnt-tracked).
__device__ __forceinline__ void async_copy_b128(void* lds_ptr, const void* gptr) {
    unsigned int ldsoff = (unsigned int)(unsigned long long)lds_ptr; // low 32b = LDS addr
    asm volatile("global_load_async_to_lds_b128 %0, %1, off"
                 :: "v"(ldsoff), "v"(gptr) : "memory");
}

// ---------------------------------------------------------------------------
// Prep: zero the output accumulator and build f16, transposed W2 for both
// branches.  w2T[o][k] = (f16) w2[c*2048 + i*64 + o]  with k = c*32 + i.
// ---------------------------------------------------------------------------
__global__ __launch_bounds__(256) void prep_kernel(
    const float* __restrict__ w2g, const float* __restrict__ w2f,
    _Float16* __restrict__ w2Tg, _Float16* __restrict__ w2Tf,
    float* __restrict__ out)
{
    int id = blockIdx.x * 256 + threadIdx.x;      // grid covers 262144 exactly
    out[id] = 0.0f;                               // NODES*OUTC == 262144
    int o = id >> 12;                             // 0..63
    int k = id & 4095;                            // 0..4095
    int c = k >> 5, i = k & 31;
    int lin = c * 2048 + i * 64 + o;
    w2Tg[(size_t)o * KDIM + k] = (_Float16)w2g[lin];
    w2Tf[(size_t)o * KDIM + k] = (_Float16)w2f[lin];
}

// ---------------------------------------------------------------------------
// KNN: one workgroup (256 threads) per query node.  Each thread scans 16
// candidates keeping a sorted local top-9, then LDS tree-merge.
// ---------------------------------------------------------------------------
template <int DIM>
__global__ __launch_bounds__(256) void knn_kernel(
    const float* __restrict__ f, int* __restrict__ idxOut)
{
    __shared__ float sd[256 * 9];
    __shared__ int   si[256 * 9];
    const int node = blockIdx.x;
    const int t    = threadIdx.x;

    float q[DIM];
#pragma unroll
    for (int d = 0; d < DIM; ++d) q[d] = f[node * DIM + d];

    float bd[9]; int bi[9];
#pragma unroll
    for (int j = 0; j < 9; ++j) { bd[j] = 3.0e38f; bi[j] = 0; }

    const int per = NODES / 256;                  // 16 candidates per thread
    for (int cc = 0; cc < per; ++cc) {
        int cand = t * per + cc;
        float d = 0.0f;
#pragma unroll
        for (int dd = 0; dd < DIM; ++dd) {
            float df = q[dd] - f[cand * DIM + dd];
            d += df * df;
        }
        float nd = d; int ni = cand;
#pragma unroll
        for (int j = 0; j < 9; ++j) {             // sorted insertion (swap chain)
            if (nd < bd[j]) {
                float td = bd[j]; bd[j] = nd; nd = td;
                int   ti = bi[j]; bi[j] = ni; ni = ti;
            }
        }
    }
#pragma unroll
    for (int j = 0; j < 9; ++j) { sd[t * 9 + j] = bd[j]; si[t * 9 + j] = bi[j]; }

    for (int s = 128; s >= 1; s >>= 1) {
        __syncthreads();
        if (t < s) {                              // merge lists t and t+s
            float od[9]; int oi[9];
            int a = 0, b = 0;
#pragma unroll
            for (int j = 0; j < 9; ++j) {
                float da = sd[t * 9 + a],        db = sd[(t + s) * 9 + b];
                int   ia = si[t * 9 + a],        ib = si[(t + s) * 9 + b];
                bool ta = (da < db) || (da == db && ia <= ib);
                if (ta) { od[j] = da; oi[j] = ia; ++a; }
                else    { od[j] = db; oi[j] = ib; ++b; }
            }
#pragma unroll
            for (int j = 0; j < 9; ++j) { sd[t * 9 + j] = od[j]; si[t * 9 + j] = oi[j]; }
        }
    }
    __syncthreads();
    if (t < 9) idxOut[node * 9 + t] = si[t];
}

// ---------------------------------------------------------------------------
// Fused AGC:  msg = P @ W2T  with  P[e, c*32+i] = h[e,c] * xj[e,i].
// Per wave: 16 edges x 64 outputs, K-loop over 128 blocks of 32.
// A fragment per block = (precomputed xj fragment) * h[row, kb]  -> WMMA f16.
// B blocks double-buffered in LDS, staged via async global->LDS (ASYNCcnt):
// one s_wait_asynccnt + one barrier per K-block; fetch of block k+1 overlaps
// the WMMAs of block k.
// ---------------------------------------------------------------------------
__global__ __launch_bounds__(256) void agc_kernel(
    const float* __restrict__ x,   const float* __restrict__ pos,
    const int*   __restrict__ idx,
    const float* __restrict__ w1,  const float* __restrict__ b1,
    const _Float16* __restrict__ w2T, const float* __restrict__ b2,
    float* __restrict__ out, float scale)
{
    __shared__ _Float16 hbuf[WGWAVES][16][HID];              // 32 KB
    __shared__ float    xjbuf[WGWAVES][16][INC];             // 16 KB
    __shared__ int      srcbuf[WGWAVES][16];                 // 0.5 KB
    __shared__ __align__(64) _Float16 Bt[2][OUTC][32];       // 8 KB [buf][o][klocal]

    const int tid  = threadIdx.x;
    const int wave = tid >> 5;
    const int lane = tid & 31;
    const int row  = lane & 15;
    const int half = lane >> 4;
    const int waveBase = blockIdx.x * EPW + wave * 16;       // grid*128 == EDGES

    // cooperative async-stage of B block 0 while we do the edge setup
    const int bo    = tid >> 2;                              // 0..63
    const int bkoff = (tid & 3) * 8;                         // 8 f16 = 16 B
    async_copy_b128(&Bt[0][bo][bkoff], w2T + (size_t)bo * KDIM + bkoff);

    if (lane < 16) srcbuf[wave][lane] = idx[waveBase + lane];
    __syncthreads();

    // ---- spherical edge attr + h = tanh(attr@w1+b1), 2 lanes per row ----
    const int e   = waveBase + row;
    const int n   = e / KNN;
    const int src = srcbuf[wave][row];
    float dx = pos[src * 3 + 0] - pos[n * 3 + 0];
    float dy = pos[src * 3 + 1] - pos[n * 3 + 1];
    float dz = pos[src * 3 + 2] - pos[n * 3 + 2];
    float rho = sqrtf(dx * dx + dy * dy + dz * dz);
    float th  = atan2f(dy, dx);
    float cz  = dz / fmaxf(rho, 1e-12f);
    float ph  = acosf(fminf(fmaxf(cz, -1.0f), 1.0f));

    for (int cc = 0; cc < 64; ++cc) {                        // lane: 64 h cols
        int c = half * 64 + cc;
        float v = rho * w1[c] + th * w1[HID + c] + ph * w1[2 * HID + c] + b1[c];
        hbuf[wave][row][c] = (_Float16)tanhf(v);
    }
    for (int ii = 0; ii < 16; ++ii) {                        // lane: 16 xj vals
        int i = half * 16 + ii;
        xjbuf[wave][row][i] = x[src * INC + i];
    }

    // ---- xj A-fragment (constant across all 128 K-blocks) ----
    // A k order used here (contiguous): element e -> K = e + 16*halfLane
    v16h xjfrag;
#pragma unroll
    for (int ei = 0; ei < 16; ++ei) {
        int i = ei + 16 * half;
        xjfrag[ei] = (_Float16)x[src * INC + i];
    }

    v8f acc[4] = {v8f{}, v8f{}, v8f{}, v8f{}};

    for (int kb8 = 0; kb8 < HID / 8; ++kb8) {
        // 8 h scalars for this wave-row in one LDS load
        v8h hchunk = *(const v8h*)(&hbuf[wave][row][kb8 * 8]);
#pragma unroll
        for (int t8 = 0; t8 < 8; ++t8) {
            const int kb  = kb8 * 8 + t8;
            const int cur = kb & 1;

            wait_asynccnt0();                    // my async write of Bt[cur] done
            __syncthreads();                     // all writes visible; prev readers done

            if (kb + 1 < HID)                    // prefetch next block into other buffer
                async_copy_b128(&Bt[cur ^ 1][bo][bkoff],
                                w2T + (size_t)bo * KDIM + (kb + 1) * 32 + bkoff);

            _Float16 hv = hchunk[t8];
            v16h hs;
#pragma unroll
            for (int qq = 0; qq < 16; ++qq) hs[qq] = hv;
            v16h afrag = xjfrag * hs;            // diag(h)·xj block

#pragma unroll
            for (int j = 0; j < 4; ++j) {
                // B k order used here (same as A): lane = n, element e -> K = e + 16*halfLane
                int o = j * 16 + row;
                v16h bfrag = *(const v16h*)(&Bt[cur][o][half * 16]);
                acc[j] = __builtin_amdgcn_wmma_f32_16x16x32_f16(
                    false, afrag, false, bfrag, (short)0, acc[j], false, false);
            }
        }
    }

    // ---- epilogue: + xj·b2 bias, scale, atomic accumulate ----
    const int col = lane & 15;
#pragma unroll
    for (int j = 0; j < 4; ++j) {
        int o = j * 16 + col;
#pragma unroll
        for (int r = 0; r < 8; ++r) {
            int rr = r + 8 * half;                           // C layout row
            int ee = waveBase + rr;
            int nn = ee / KNN;
            float bias = 0.0f;
            for (int i = 0; i < INC; ++i)
                bias += xjbuf[wave][rr][i] * b2[i * OUTC + o];
            atomicAdd(&out[nn * OUTC + o], (acc[j][r] + bias) * scale);
        }
    }
}

// ---------------------------------------------------------------------------
extern "C" void kernel_launch(void* const* d_in, const int* in_sizes, int n_in,
                              void* d_out, int out_size, void* d_ws, size_t ws_size,
                              hipStream_t stream) {
    const float* x   = (const float*)d_in[0];
    const float* pos = (const float*)d_in[1];
    const float* w1g = (const float*)d_in[2];
    const float* b1g = (const float*)d_in[3];
    const float* w2g = (const float*)d_in[4];
    const float* b2g = (const float*)d_in[5];
    const float* w1f = (const float*)d_in[6];
    const float* b1f = (const float*)d_in[7];
    const float* w2f = (const float*)d_in[8];
    const float* b2f = (const float*)d_in[9];
    float* out = (float*)d_out;

    char* ws = (char*)d_ws;
    int*      idx_geo  = (int*)ws;                            // 147456 B
    int*      idx_feat = (int*)(ws + 147456);                 // 147456 B
    _Float16* w2Tg     = (_Float16*)(ws + 294912);            // 524288 B
    _Float16* w2Tf     = (_Float16*)(ws + 819200);            // 524288 B

    prep_kernel<<<1024, 256, 0, stream>>>(w2g, w2f, w2Tg, w2Tf, out);
    knn_kernel<3><<<NODES, 256, 0, stream>>>(pos, idx_geo);
    knn_kernel<32><<<NODES, 256, 0, stream>>>(x, idx_feat);

    const float scale = 0.5f / (float)KNN;                    // mean over K, avg branches
    agc_kernel<<<EDGES / EPW, 256, 0, stream>>>(x, pos, idx_geo,
                                                w1g, b1g, w2Tg, b2g, out, scale);
    agc_kernel<<<EDGES / EPW, 256, 0, stream>>>(x, pos, idx_feat,
                                                w1f, b1f, w2Tf, b2f, out, scale);
}
